// XDomainMixEMG_44461501448580
// MI455X (gfx1250) — compile-verified, matrix operand order checked by reading, not run on hardware
//
#include <hip/hip_runtime.h>
#include <math.h>

// ---------------------------------------------------------------------------
// XDomainMixEMG on MI455X (gfx1250, wave32, WMMA)
//
// Roofline: swap_scores argmax reads 1 GiB (dominant, ~46us @ 23.3 TB/s).
// GEMMs total < 4 GFLOP -> run them on the bf16 WMMA pipe; everything fits L2.
// Weights are pre-packed TRANSPOSED (N x K) in bf16 so both A and B fragments
// are two contiguous 128-bit loads per lane (matches ISA 7.12.2 16-bit layout:
// lane&15 selects row/col, lane>>4 selects K-base {0,8}; VGPR0..3=K+0..7,
// VGPR4..7=K+16..23).
// ---------------------------------------------------------------------------

#define BN     16384
#define SHARED 256
#define CG 32
#define CS 96
#define DG 32
#define DS 64
#define NG 10
#define NSUB 4

typedef __attribute__((ext_vector_type(16))) __bf16 v16bf;
typedef __attribute__((ext_vector_type(8)))  __bf16 v8bf;
typedef __attribute__((ext_vector_type(8)))  float  v8f;

union V16U { v16bf v; v8bf h[2]; };

// ---------------------------------------------------------------------------
// bf16 WMMA GEMM, one wave per 64x16 output tile (4 accumulators share one
// B fragment per K-step). K is a template constant -> fully unrolled loop.
// A:  row-major M x K bf16 (lda)
// Wt: TRANSPOSED weights, row-major Npad x K bf16 (column n of W at Wt+n*K)
// Epilogue: +bias[col], optional ReLU, optional fp32 store (col < Nvalid),
// up to two bf16 stores (builds concatenated activation matrices in-place).
// ---------------------------------------------------------------------------
template <int K>
__global__ void __launch_bounds__(32)
gemm_bf16_wmma(const __bf16* __restrict__ A, int lda,
               const __bf16* __restrict__ Wt,
               const float* __restrict__ bias,
               float* __restrict__ Cf, int ldc, int Nvalid,
               __bf16* __restrict__ Cb, int ldcb,
               __bf16* __restrict__ Cb2, int ldcb2,
               int relu)
{
    const int lane = threadIdx.x;
    const int r    = lane & 15;
    const int half = lane >> 4;
    const int kb   = half * 8;
    const int tm   = blockIdx.x * 64;
    const int tn   = blockIdx.y * 16;

    v8f acc[4] = {};

    const __bf16* Brow = Wt + (tn + r) * K + kb;        // contiguous in K
    const __bf16* Arow = A + (tm + r) * lda + kb;

#pragma unroll
    for (int k0 = 0; k0 < K; k0 += 32) {
        V16U b;
        b.h[0] = *reinterpret_cast<const v8bf*>(Brow + k0);
        b.h[1] = *reinterpret_cast<const v8bf*>(Brow + k0 + 16);
#pragma unroll
        for (int t = 0; t < 4; ++t) {
            V16U a;
            const __bf16* ap = Arow + t * 16 * lda + k0;
            a.h[0] = *reinterpret_cast<const v8bf*>(ap);
            a.h[1] = *reinterpret_cast<const v8bf*>(ap + 16);
            acc[t] = __builtin_amdgcn_wmma_f32_16x16x32_bf16(
                /*neg_a=*/false, a.v, /*neg_b=*/false, b.v,
                /*c_mod=*/(short)0, acc[t], /*reuse_a=*/false, /*reuse_b=*/false);
        }
    }

    const int col = tn + r;
    const float bv = bias ? bias[col] : 0.0f;
#pragma unroll
    for (int t = 0; t < 4; ++t) {
#pragma unroll
        for (int v = 0; v < 8; ++v) {
            const int row = tm + t * 16 + v + half * 8;
            float x = acc[t][v] + bv;
            if (relu) x = fmaxf(x, 0.0f);
            if (Cf && col < Nvalid) Cf[(size_t)row * ldc + col] = x;
            if (Cb)                 Cb [(size_t)row * ldcb  + col] = (__bf16)x;
            if (Cb2)                Cb2[(size_t)row * ldcb2 + col] = (__bf16)x;
        }
    }
}

// ---------------------------------------------------------------------------
// Masked row argmax over the 1 GiB swap_scores matrix. One 256-thread block
// per row; float4/int4 coalesced streaming; first-occurrence tie-breaking to
// match jnp.argmax semantics; fallback to identity when no valid column.
// ---------------------------------------------------------------------------
__global__ void __launch_bounds__(256)
argmax_rows(const float* __restrict__ scores, const int* __restrict__ labels,
            int* __restrict__ perm)
{
    const int i   = blockIdx.x;
    const int tid = threadIdx.x;
    const int myLab = labels[i];
    const float4* row4 = reinterpret_cast<const float4*>(scores + (size_t)i * BN);
    const int4*   lab4 = reinterpret_cast<const int4*>(labels);

    float best = -INFINITY;
    int   bidx = -1;
    const int nVec = BN >> 2;
    for (int v = tid; v < nVec; v += 256) {
        if (v + 256 < nVec) __builtin_prefetch(row4 + v + 256, 0, 1);
        float4 s = row4[v];
        int4   l = lab4[v];
        const int j = v << 2;
        if (l.x != myLab && s.x > best) { best = s.x; bidx = j; }
        if (l.y != myLab && s.y > best) { best = s.y; bidx = j + 1; }
        if (l.z != myLab && s.z > best) { best = s.z; bidx = j + 2; }
        if (l.w != myLab && s.w > best) { best = s.w; bidx = j + 3; }
    }

    __shared__ float sv[256];
    __shared__ int   si[256];
    sv[tid] = best; si[tid] = bidx;
    __syncthreads();
    for (int s = 128; s > 0; s >>= 1) {
        if (tid < s) {
            const float ov = sv[tid + s]; const int oi = si[tid + s];
            if (ov > sv[tid] ||
                (ov == sv[tid] && oi >= 0 && (si[tid] < 0 || oi < si[tid]))) {
                sv[tid] = ov; si[tid] = oi;
            }
        }
        __syncthreads();
    }
    if (tid == 0) perm[i] = (si[0] < 0) ? i : si[0];
}

// ---------------------------------------------------------------------------
// Small helpers
// ---------------------------------------------------------------------------
__global__ void f32_to_bf16(const float* __restrict__ src,
                            __bf16* __restrict__ dst, size_t n)
{
    size_t idx = (size_t)blockIdx.x * blockDim.x + threadIdx.x;
    if (idx < n) dst[idx] = (__bf16)src[idx];
}

// transpose + convert + zero-pad: W (K x N) fp32 -> Wt (Npad x K) bf16
__global__ void pack_weight_t(const float* __restrict__ W, __bf16* __restrict__ out,
                              int K, int N, int Npad)
{
    int idx = blockIdx.x * blockDim.x + threadIdx.x;
    if (idx < Npad * K) {
        int n = idx / K, k = idx - n * K;
        out[idx] = (__bf16)((n < N) ? W[(size_t)k * N + n] : 0.0f);
    }
}

// bias pad (+optional constant fold, used for FiLM's 1+gamma)
__global__ void pack_bias(const float* __restrict__ b, float* __restrict__ out,
                          int N, int Npad, float addc)
{
    int j = blockIdx.x * blockDim.x + threadIdx.x;
    if (j < Npad) out[j] = (j < N) ? (b[j] + addc) : 0.0f;
}

// z_ds_swap = z_ds[perm]; fp32 to d_out + bf16 copy for FiLM GEMM input
__global__ void __launch_bounds__(64)
gather_zds(const float* __restrict__ zds, const int* __restrict__ perm,
           float* __restrict__ outf, __bf16* __restrict__ outb)
{
    const int i = blockIdx.x, t = threadIdx.x;
    const int p = perm[i];
    const float v = zds[(size_t)p * DS + t];
    outf[(size_t)i * DS + t] = v;
    outb[(size_t)i * DS + t] = (__bf16)v;
}

// X_aug[:, 0:96] = bf16(gamma * z_cs + beta)   (gamma already includes +1)
__global__ void film_apply(const float* __restrict__ gamma,
                           const float* __restrict__ beta,
                           const float* __restrict__ zcs,
                           __bf16* __restrict__ xaug, int n)
{
    int idx = blockIdx.x * blockDim.x + threadIdx.x;
    if (idx < n) {
        int i = idx / CS, j = idx - i * CS;
        float v = gamma[idx] * zcs[idx] + beta[idx];
        xaug[(size_t)i * (CS + CG) + j] = (__bf16)v;
    }
}

// ---------------------------------------------------------------------------
extern "C" void kernel_launch(void* const* d_in, const int* in_sizes, int n_in,
                              void* d_out, int out_size, void* d_ws, size_t ws_size,
                              hipStream_t stream)
{
    (void)in_sizes; (void)n_in; (void)out_size; (void)ws_size;

    // ---- inputs (setup_inputs order) ----
    const float* shared  = (const float*)d_in[0];
    const int*   labels  = (const int*)  d_in[1];
    const float* scores  = (const float*)d_in[2];
    const float* W_cg = (const float*)d_in[3];  const float* b_cg = (const float*)d_in[4];
    const float* W_cs = (const float*)d_in[5];  const float* b_cs = (const float*)d_in[6];
    const float* W_dg = (const float*)d_in[7];  const float* b_dg = (const float*)d_in[8];
    const float* W_ds = (const float*)d_in[9];  const float* b_ds = (const float*)d_in[10];
    const float* fgW  = (const float*)d_in[11]; const float* fgb  = (const float*)d_in[12];
    const float* fbW  = (const float*)d_in[13]; const float* fbb  = (const float*)d_in[14];
    const float* g1W  = (const float*)d_in[15]; const float* g1b  = (const float*)d_in[16];
    const float* g2W  = (const float*)d_in[17]; const float* g2b  = (const float*)d_in[18];
    const float* d1W  = (const float*)d_in[19]; const float* d1b  = (const float*)d_in[20];
    const float* d2W  = (const float*)d_in[21]; const float* d2b  = (const float*)d_in[22];

    // ---- outputs (tuple order, concatenated fp32) ----
    float* out = (float*)d_out;
    float* o_glb = out;                         // (B, 10)
    float* o_gla = o_glb + (size_t)BN * NG;     // (B, 10)
    float* o_dl  = o_gla + (size_t)BN * NG;     // (B, 4)
    float* o_zcg = o_dl  + (size_t)BN * NSUB;   // (B, 32)
    float* o_zcs = o_zcg + (size_t)BN * CG;     // (B, 96)
    float* o_zdg = o_zcs + (size_t)BN * CS;     // (B, 32)
    float* o_zds = o_zdg + (size_t)BN * DG;     // (B, 64)
    float* o_zsw = o_zds + (size_t)BN * DS;     // (B, 64)

    // ---- workspace carve-up ----
    char* ws = (char*)d_ws;
    size_t off = 0;
    auto take = [&](size_t bytes) { char* p = ws + off; off = (off + bytes + 255) & ~(size_t)255; return p; };
    __bf16* sharedBF = (__bf16*)take((size_t)BN * SHARED * 2);   // 8 MiB
    __bf16* X_base   = (__bf16*)take((size_t)BN * 128 * 2);      // [z_cs | z_cg]
    __bf16* X_aug    = (__bf16*)take((size_t)BN * 128 * 2);      // [film | z_cg]
    __bf16* X_dom    = (__bf16*)take((size_t)BN * 128 * 2);      // [z_ds | z_dg]
    __bf16* Zswap_bf = (__bf16*)take((size_t)BN * DS * 2);
    __bf16* H_base   = (__bf16*)take((size_t)BN * 64 * 2);
    __bf16* H_aug    = (__bf16*)take((size_t)BN * 64 * 2);
    __bf16* H_dom    = (__bf16*)take((size_t)BN * 128 * 2);
    float*  gammaB   = (float*) take((size_t)BN * CS * 4);
    float*  betaB    = (float*) take((size_t)BN * CS * 4);
    int*    perm     = (int*)   take((size_t)BN * 4);
    __bf16* Wcg_bf = (__bf16*)take((size_t)CG * SHARED * 2);     // transposed N x K
    __bf16* Wcs_bf = (__bf16*)take((size_t)CS * SHARED * 2);
    __bf16* Wdg_bf = (__bf16*)take((size_t)DG * SHARED * 2);
    __bf16* Wds_bf = (__bf16*)take((size_t)DS * SHARED * 2);
    __bf16* fgW_bf = (__bf16*)take((size_t)CS * DS * 2);
    __bf16* fbW_bf = (__bf16*)take((size_t)CS * DS * 2);
    __bf16* g1W_bf = (__bf16*)take((size_t)64 * 128 * 2);
    __bf16* g2W_bf = (__bf16*)take((size_t)16 * 64 * 2);         // N=10 -> pad 16
    __bf16* d1W_bf = (__bf16*)take((size_t)128 * 128 * 2);
    __bf16* d2W_bf = (__bf16*)take((size_t)16 * 128 * 2);        // N=4  -> pad 16
    float*  fgb1   = (float*) take(CS * 4);                      // fgb + 1.0
    float*  g2b_p  = (float*) take(16 * 4);
    float*  d2b_p  = (float*) take(16 * 4);

    // ---- 1) masked argmax over swap_scores (the 1 GiB read; dominant) ----
    argmax_rows<<<BN, 256, 0, stream>>>(scores, labels, perm);

    // ---- 2) fp32 -> bf16 staging ----
    {
        size_t n = (size_t)BN * SHARED;
        f32_to_bf16<<<(unsigned)((n + 255) / 256), 256, 0, stream>>>(shared, sharedBF, n);
    }
    auto packW = [&](const float* Wsrc, __bf16* Wdst, int K, int N, int Npad) {
        int tot = Npad * K;
        pack_weight_t<<<(tot + 255) / 256, 256, 0, stream>>>(Wsrc, Wdst, K, N, Npad);
    };
    packW(W_cg, Wcg_bf, SHARED, CG, CG);
    packW(W_cs, Wcs_bf, SHARED, CS, CS);
    packW(W_dg, Wdg_bf, SHARED, DG, DG);
    packW(W_ds, Wds_bf, SHARED, DS, DS);
    packW(fgW,  fgW_bf, DS, CS, CS);
    packW(fbW,  fbW_bf, DS, CS, CS);
    packW(g1W,  g1W_bf, 128, 64, 64);
    packW(g2W,  g2W_bf, 64,  NG, 16);
    packW(d1W,  d1W_bf, 128, 128, 128);
    packW(d2W,  d2W_bf, 128, NSUB, 16);
    pack_bias<<<1, 128, 0, stream>>>(fgb, fgb1, CS, CS, 1.0f);   // fold the FiLM +1
    pack_bias<<<1, 32, 0, stream>>>(g2b, g2b_p, NG, 16, 0.0f);
    pack_bias<<<1, 32, 0, stream>>>(d2b, d2b_p, NSUB, 16, 0.0f);

    // ---- GEMM dispatcher (K is compile-time -> fully unrolled WMMA loop) ----
    auto gemm = [&](const __bf16* A, int lda, const __bf16* Wt,
                    const float* bias, float* Cf, int ldc, int Nvalid,
                    __bf16* Cb, int ldcb, __bf16* Cb2, int ldcb2,
                    int M, int Npad, int K, int relu) {
        dim3 grid(M / 64, Npad / 16);
        switch (K) {
        case 64:
            gemm_bf16_wmma<64><<<grid, 32, 0, stream>>>(A, lda, Wt, bias,
                Cf, ldc, Nvalid, Cb, ldcb, Cb2, ldcb2, relu); break;
        case 128:
            gemm_bf16_wmma<128><<<grid, 32, 0, stream>>>(A, lda, Wt, bias,
                Cf, ldc, Nvalid, Cb, ldcb, Cb2, ldcb2, relu); break;
        default:
            gemm_bf16_wmma<256><<<grid, 32, 0, stream>>>(A, lda, Wt, bias,
                Cf, ldc, Nvalid, Cb, ldcb, Cb2, ldcb2, relu); break;
        }
    };

    // ---- 3) projection GEMMs (bf16 WMMA), fused epilogue builds concat bufs ----
    // z_cs -> out + X_base[:,0:96]
    gemm(sharedBF, SHARED, Wcs_bf, b_cs, o_zcs, CS, CS,
         X_base, 128, nullptr, 0, BN, CS, SHARED, 0);
    // z_cg -> out + X_base[:,96:128] + X_aug[:,96:128]
    gemm(sharedBF, SHARED, Wcg_bf, b_cg, o_zcg, CG, CG,
         X_base + CS, 128, X_aug + CS, 128, BN, CG, SHARED, 0);
    // z_ds -> out + X_dom[:,0:64]
    gemm(sharedBF, SHARED, Wds_bf, b_ds, o_zds, DS, DS,
         X_dom, 128, nullptr, 0, BN, DS, SHARED, 0);
    // z_dg -> out + X_dom[:,64:96]
    gemm(sharedBF, SHARED, Wdg_bf, b_dg, o_zdg, DG, DG,
         X_dom + DS, 128, nullptr, 0, BN, DG, SHARED, 0);

    // ---- 4) swap gather (needs perm + z_ds) ----
    gather_zds<<<BN, 64, 0, stream>>>(o_zds, perm, o_zsw, Zswap_bf);

    // ---- 5) FiLM: gamma = 1 + Zswap@fgW + fgb ; beta = Zswap@fbW + fbb ----
    gemm(Zswap_bf, DS, fgW_bf, fgb1, gammaB, CS, CS,
         nullptr, 0, nullptr, 0, BN, CS, DS, 0);
    gemm(Zswap_bf, DS, fbW_bf, fbb, betaB, CS, CS,
         nullptr, 0, nullptr, 0, BN, CS, DS, 0);
    film_apply<<<(BN * CS + 255) / 256, 256, 0, stream>>>(gammaB, betaB, o_zcs,
                                                          X_aug, BN * CS);

    // ---- 6) gesture MLPs (base + aug) ----
    gemm(X_base, 128, g1W_bf, g1b, nullptr, 0, 0,
         H_base, 64, nullptr, 0, BN, 64, 128, 1);
    gemm(H_base, 64, g2W_bf, g2b_p, o_glb, NG, NG,
         nullptr, 0, nullptr, 0, BN, 16, 64, 0);
    gemm(X_aug, 128, g1W_bf, g1b, nullptr, 0, 0,
         H_aug, 64, nullptr, 0, BN, 64, 128, 1);
    gemm(H_aug, 64, g2W_bf, g2b_p, o_gla, NG, NG,
         nullptr, 0, nullptr, 0, BN, 16, 64, 0);

    // ---- 7) domain MLP ----
    gemm(X_dom, 128, d1W_bf, d1b, nullptr, 0, 0,
         H_dom, 128, nullptr, 0, BN, 128, 128, 1);
    gemm(H_dom, 128, d2W_bf, d2b_p, o_dl, NSUB, NSUB,
         nullptr, 0, nullptr, 0, BN, 16, 128, 0);
}
